// DocumentRQVAE_76613626626056
// MI455X (gfx1250) — matrix operand order repaired and checked by reading, not compiled, not run on hardware
//
#include <hip/hip_runtime.h>
#include <stdint.h>

typedef __attribute__((ext_vector_type(16))) _Float16 v16h;
typedef __attribute__((ext_vector_type(8)))  float    v8f;
typedef int v4i __attribute__((vector_size(16)));   // matches builtin param type

// ---------------------------------------------------------------------------
// CDNA5 async global->LDS copy (ASYNCcnt) with sync fallback.
// Builtin signature (from hipcc diagnostic): param0 = 'int __vector(4) __device__ *'
// i.e. v4i in address space 1; param1 = LDS-side v4i pointer; two immediates.
// Generic LDS ptr low 32 bits == LDS offset (aperture rule); generic global
// ptr value == AS1 address, so integer casts build the AS-qualified pointers.
// ---------------------------------------------------------------------------
#if defined(__HIP_DEVICE_COMPILE__) && __has_builtin(__builtin_amdgcn_global_load_async_to_lds_b128)
#define GL_ASYNC_OK 1
#else
#define GL_ASYNC_OK 0
#endif

__device__ __forceinline__ void async_copy_b128(const _Float16* g, _Float16* l) {
#if GL_ASYNC_OK
    __builtin_amdgcn_global_load_async_to_lds_b128(
        (__attribute__((address_space(1))) v4i*)(uintptr_t)g,
        (__attribute__((address_space(3))) v4i*)(uint32_t)(uintptr_t)l,
        0, 0);
#else
    *(float4*)l = *(const float4*)g;
#endif
}

#if GL_ASYNC_OK
#if __has_builtin(__builtin_amdgcn_s_wait_asynccnt)
#define WAIT_ASYNC0() __builtin_amdgcn_s_wait_asynccnt(0)
#else
#define WAIT_ASYNC0() asm volatile("s_wait_asynccnt 0x0" ::: "memory")
#endif
#else
#define WAIT_ASYNC0() do {} while (0)
#endif

// ---------------------------------------------------------------------------
// WMMA GEMM:  C[M,N] = epilogue( A[M,K] (f16) @ Bt[N,K]^T (f16) ), f32 accum.
// Double-buffered LDS, async staging.
// EPI: 0 = +bias -> Cf            (decoder output projection)
//      1 = relu(LN(+bias))        (first resblock matmul)      [BN==N]
//      2 = Hres(f16) + LN(+bias)  (second resblock matmul)     [BN==N]
//      3 = tanh(+bias) -> Cf      (attention pooling hidden)
//      4 = bias[n] - 2*acc -> Cf  (VQ distance, bias = codebook row norms)
// LN epilogues write Cf (f32) and/or Ch (f16) if non-null.
// ---------------------------------------------------------------------------
template<int BM, int BN, int WROWS, int WCOLS, int EPI>
__global__ void __launch_bounds__(256) gemm_f16(
    const _Float16* __restrict__ A, const _Float16* __restrict__ Bt,
    const float* __restrict__ bias, const float* __restrict__ gamma,
    const float* __restrict__ beta, const _Float16* __restrict__ Hres,
    float* __restrict__ Cf, _Float16* __restrict__ Ch, int M, int N, int K)
{
    constexpr int KT   = 32;
    constexpr int LSTR = KT + 8;                 // halves; rows 16B aligned
    constexpr int MT   = BM / (WROWS * 16);
    constexpr int NT   = BN / (WCOLS * 16);
    constexpr bool LN  = (EPI == 1 || EPI == 2);
    constexpr int ABUF = BM * LSTR;              // halves per A buffer
    constexpr int BBUF = BN * LSTR;
    constexpr unsigned TILEB = 2u * (unsigned)(ABUF + BBUF) * 2u;  // double buffered
    constexpr unsigned LNB   = (unsigned)BM * BN * 4u + (unsigned)BM * 8u;
    constexpr unsigned SMEMB = LN ? (TILEB > LNB ? TILEB : LNB) : TILEB;

    __shared__ alignas(16) unsigned char smem[SMEMB];
    _Float16* sA = (_Float16*)smem;              // [2][BM][LSTR]
    _Float16* sB = sA + 2 * ABUF;                // [2][BN][LSTR]

    const int tid  = threadIdx.x;
    const int wave = tid >> 5;
    const int lane = tid & 31;
    const int lrow = lane & 15;
    const int lhi  = lane >> 4;
    const int wr   = wave / WCOLS;
    const int wc   = wave % WCOLS;
    const int bm0  = blockIdx.y * BM;
    const int bn0  = blockIdx.x * BN;

    auto stage = [&](int k0, int sel) {
        _Float16* dA = sA + sel * ABUF;
        _Float16* dB = sB + sel * BBUF;
        for (int c = tid; c < (BM * KT) / 8; c += 256) {
            int rr = c >> 2, c8 = (c & 3) * 8;   // KT/8 == 4 chunks per row
            async_copy_b128(A + (size_t)(bm0 + rr) * K + k0 + c8, dA + rr * LSTR + c8);
        }
        for (int c = tid; c < (BN * KT) / 8; c += 256) {
            int rr = c >> 2, c8 = (c & 3) * 8;
            async_copy_b128(Bt + (size_t)(bn0 + rr) * K + k0 + c8, dB + rr * LSTR + c8);
        }
    };

    v8f acc[MT][NT];
    const v8f vzero = {0.f,0.f,0.f,0.f,0.f,0.f,0.f,0.f};
#pragma unroll
    for (int m = 0; m < MT; m++)
#pragma unroll
        for (int n = 0; n < NT; n++) acc[m][n] = vzero;

    const int nst = K / KT;
    stage(0, 0);
    for (int t = 0; t < nst; t++) {
        const int cur = t & 1;
        WAIT_ASYNC0();                 // tile t landed (this wave's portion)
        __syncthreads();               // all waves' portions visible
        if (t + 1 < nst) stage((t + 1) * KT, 1 - cur);   // prefetch next tile

        const _Float16* cA = sA + cur * ABUF;
        const _Float16* cB = sB + cur * BBUF;
        const int kb = lhi * 8;
        v16h af[MT];
#pragma unroll
        for (int m = 0; m < MT; m++) {
            const _Float16* p = cA + (wr * MT * 16 + m * 16 + lrow) * LSTR;
#pragma unroll
            for (int j = 0; j < 8; j++) {
                af[m][j]     = p[kb + j];            // K = lhi*8 + j
                af[m][8 + j] = p[16 + kb + j];       // K = 16 + lhi*8 + j
            }
        }
        v16h bf[NT];
#pragma unroll
        for (int n = 0; n < NT; n++) {
            const _Float16* p = cB + (wc * NT * 16 + n * 16 + lrow) * LSTR;
#pragma unroll
            for (int j = 0; j < 8; j++) {
                bf[n][j]     = p[kb + j];            // K = lhi*8 + j
                bf[n][8 + j] = p[16 + kb + j];       // K = 16 + lhi*8 + j
            }
        }
#pragma unroll
        for (int m = 0; m < MT; m++)
#pragma unroll
            for (int n = 0; n < NT; n++)
                acc[m][n] = __builtin_amdgcn_wmma_f32_16x16x32_f16(
                    false, af[m], false, bf[n], (short)0, acc[m][n], false, false);
    }

    if constexpr (!LN) {
#pragma unroll
        for (int m = 0; m < MT; m++)
#pragma unroll
            for (int n = 0; n < NT; n++)
#pragma unroll
                for (int g = 0; g < 8; g++) {
                    int rl = wr * MT * 16 + m * 16 + g + 8 * lhi;
                    int cg = bn0 + wc * NT * 16 + n * 16 + lrow;
                    float v = acc[m][n][g];
                    if constexpr (EPI == 0) v = v + bias[cg];
                    if constexpr (EPI == 3) v = tanhf(v + bias[cg]);
                    if constexpr (EPI == 4) v = bias[cg] - 2.0f * v;
                    Cf[(size_t)(bm0 + rl) * N + cg] = v;
                }
    } else {
        float* sC    = (float*)smem;
        float* sStat = (float*)(smem + (unsigned)BM * BN * 4u);
        __syncthreads();               // everyone done reading tile buffers
#pragma unroll
        for (int m = 0; m < MT; m++)
#pragma unroll
            for (int n = 0; n < NT; n++)
#pragma unroll
                for (int g = 0; g < 8; g++) {
                    int rl = wr * MT * 16 + m * 16 + g + 8 * lhi;
                    int cl = wc * NT * 16 + n * 16 + lrow;
                    sC[rl * BN + cl] = acc[m][n][g] + bias[cl];
                }
        __syncthreads();
        if (tid < BM) {
            float s = 0.f, s2 = 0.f;
            for (int c2 = 0; c2 < BN; c2++) { float v = sC[tid * BN + c2]; s += v; s2 += v * v; }
            float mean = s / BN;
            float var  = s2 / BN - mean * mean;
            sStat[tid * 2]     = mean;
            sStat[tid * 2 + 1] = rsqrtf(var + 1e-5f);
        }
        __syncthreads();
        for (int e = tid; e < BM * BN; e += 256) {
            int rl = e / BN, cl = e % BN;
            float v = (sC[e] - sStat[rl * 2]) * sStat[rl * 2 + 1] * gamma[cl] + beta[cl];
            if constexpr (EPI == 1) v = fmaxf(v, 0.f);
            if constexpr (EPI == 2) v += (float)Hres[(size_t)(bm0 + rl) * N + cl];
            size_t idx = (size_t)(bm0 + rl) * N + cl;
            if (Cf) Cf[idx] = v;
            if (Ch) Ch[idx] = (_Float16)v;
        }
    }
}

// ---------------------------------------------------------------------------
// Prep kernels
// ---------------------------------------------------------------------------
__global__ void transpose_cvt(const float* __restrict__ src, _Float16* __restrict__ dst,
                              int K, int N) {   // src[K][N] -> dst[N][K]
    size_t i = (size_t)blockIdx.x * 256 + threadIdx.x;
    if (i < (size_t)K * N) {
        int k = (int)(i / N), n = (int)(i % N);
        dst[(size_t)n * K + k] = (_Float16)src[i];
    }
}
__global__ void cvt_f16_vec(const float* __restrict__ src, _Float16* __restrict__ dst,
                            size_t n4) {        // n4 = elements/4
    size_t i = (size_t)blockIdx.x * 256 + threadIdx.x;
    if (i < n4) {
        float4 v = ((const float4*)src)[i];
        _Float16* d = dst + i * 4;
        d[0] = (_Float16)v.x; d[1] = (_Float16)v.y;
        d[2] = (_Float16)v.z; d[3] = (_Float16)v.w;
    }
}
__global__ void cb_norms(const float* __restrict__ cb, float* __restrict__ norms) {
    int i = blockIdx.x * 256 + threadIdx.x;     // 4*2048 rows
    if (i < 4 * 2048) {
        const float* e = cb + (size_t)i * 256;
        float s = 0.f;
        for (int d = 0; d < 256; d++) s += e[d] * e[d];
        norms[i] = s;
    }
}

// ---------------------------------------------------------------------------
// Pooling
// ---------------------------------------------------------------------------
__global__ void score_softmax(const float* __restrict__ scoresH, const float* __restrict__ w2,
                              const float* __restrict__ b2, const float* __restrict__ mask,
                              float* __restrict__ wout) {
    int b = blockIdx.x, s = threadIdx.x;        // 128 threads
    const float* row = scoresH + ((size_t)b * 128 + s) * 128;
    float sc = b2[0];
    for (int j = 0; j < 128; j++) sc += row[j] * w2[j];
    float m = mask[b * 128 + s];
    if (m == 0.0f) sc = -__builtin_inff();
    __shared__ float red[128];
    red[s] = sc; __syncthreads();
    for (int k = 64; k > 0; k >>= 1) { if (s < k) red[s] = fmaxf(red[s], red[s + k]); __syncthreads(); }
    float mx = red[0]; __syncthreads();
    float e = (m == 0.0f) ? 0.0f : __expf(sc - mx);
    red[s] = e; __syncthreads();
    for (int k = 64; k > 0; k >>= 1) { if (s < k) red[s] += red[s + k]; __syncthreads(); }
    wout[b * 128 + s] = e / red[0];
}

__global__ void pool_sum(const _Float16* __restrict__ z, const float* __restrict__ w,
                         float* __restrict__ r, _Float16* __restrict__ r16,
                         float* __restrict__ qout) {
    int b = blockIdx.x, d = threadIdx.x;        // 256 threads
    __shared__ float ws[128];
    if (d < 128) ws[d] = w[b * 128 + d];
    __syncthreads();
    float s = 0.f;
    for (int si = 0; si < 128; si++) s += (float)z[((size_t)b * 128 + si) * 256 + d] * ws[si];
    r[b * 256 + d] = s;
    r16[b * 256 + d] = (_Float16)s;
    qout[b * 256 + d] = 0.f;
}

// ---------------------------------------------------------------------------
// RVQ
// ---------------------------------------------------------------------------
__global__ void argmin_k(const float* __restrict__ Dm, int* __restrict__ idxb,
                         int* __restrict__ codes, int ib) {
    int b = blockIdx.x, t = threadIdx.x;
    float best = 3.4e38f; int bi = 0;
    for (int n = t; n < 2048; n += 256) {
        float v = Dm[(size_t)b * 2048 + n];
        if (v < best) { best = v; bi = n; }
    }
    __shared__ float sv[256]; __shared__ int si2[256];
    sv[t] = best; si2[t] = bi; __syncthreads();
    for (int k = 128; k > 0; k >>= 1) {
        if (t < k) {
            float ov = sv[t + k]; int oi = si2[t + k];
            if (ov < sv[t] || (ov == sv[t] && oi < si2[t])) { sv[t] = ov; si2[t] = oi; }
        }
        __syncthreads();
    }
    if (t == 0) { idxb[b] = si2[0]; codes[b * 4 + ib] = si2[0]; }
}

// forward: q_st == q, vq_loss_i = 1.25*mean((q-r)^2); q_out += q; r -= q
__global__ void vq_update(const float* __restrict__ cb, const int* __restrict__ idxb,
                          float* __restrict__ r, _Float16* __restrict__ r16,
                          float* __restrict__ qout, _Float16* __restrict__ qout16,
                          float* __restrict__ part) {
    int b = blockIdx.x, d = threadIdx.x;        // 256 threads
    int id = idxb[b];
    float q  = cb[(size_t)id * 256 + d];
    float rr = r[b * 256 + d];
    float df = q - rr;
    __shared__ float red[256];
    red[d] = df * df; __syncthreads();
    for (int k = 128; k > 0; k >>= 1) { if (d < k) red[d] += red[d + k]; __syncthreads(); }
    if (d == 0) part[b] = red[0];
    float nq = qout[b * 256 + d] + q;
    qout[b * 256 + d] = nq;
    qout16[b * 256 + d] = (_Float16)nq;
    float nr = rr - q;
    r[b * 256 + d] = nr;
    r16[b * 256 + d] = (_Float16)nr;
}

// ---------------------------------------------------------------------------
// Target + deterministic losses
// ---------------------------------------------------------------------------
__global__ void x_target_k(const float* __restrict__ x, const float* __restrict__ mask,
                           float* __restrict__ xt) {
    int b = blockIdx.x, t = threadIdx.x;
    float sm = 0.f;
    for (int s = 0; s < 128; s++) sm += mask[b * 128 + s];
    float inv = 1.0f / sm;
    for (int c = t; c < 768; c += 256) {
        float acc = 0.f;
        for (int s = 0; s < 128; s++)
            acc += x[((size_t)b * 128 + s) * 768 + c] * mask[b * 128 + s];
        xt[(size_t)b * 768 + c] = acc * inv;
    }
}
__global__ void recon_part_k(const float* __restrict__ xr, const float* __restrict__ xt,
                             float* __restrict__ parts) {
    int b = blockIdx.x, t = threadIdx.x;
    float s = 0.f;
    for (int c = t; c < 768; c += 256) {
        float d = xr[(size_t)b * 768 + c] - xt[(size_t)b * 768 + c];
        s += d * d;
    }
    __shared__ float red[256];
    red[t] = s; __syncthreads();
    for (int k = 128; k > 0; k >>= 1) { if (t < k) red[t] += red[t + k]; __syncthreads(); }
    if (t == 0) parts[b] = red[0];
}
__global__ void final_loss_k(const float* __restrict__ parts, const float* __restrict__ vqpart,
                             float* __restrict__ out) {
    __shared__ float red[256];
    int t = threadIdx.x;
    float s = 0.f;
    for (int i = t; i < 1024; i += 256) s += parts[i];
    red[t] = s; __syncthreads();
    for (int k = 128; k > 0; k >>= 1) { if (t < k) red[t] += red[t + k]; __syncthreads(); }
    float recon = red[0] * (1.0f / (1024.0f * 768.0f));
    __syncthreads();
    float v = 0.f;
    for (int i = t; i < 4096; i += 256) v += vqpart[i];
    red[t] = v; __syncthreads();
    for (int k = 128; k > 0; k >>= 1) { if (t < k) red[t] += red[t + k]; __syncthreads(); }
    if (t == 0) out[0] = recon + red[0] * (1.25f / (1024.0f * 256.0f));
}

// ---------------------------------------------------------------------------
extern "C" void kernel_launch(void* const* d_in, const int* in_sizes, int n_in,
                              void* d_out, int out_size, void* d_ws, size_t ws_size,
                              hipStream_t stream) {
    (void)in_sizes; (void)n_in; (void)out_size; (void)ws_size;
    const float* x      = (const float*)d_in[0];
    const float* mask   = (const float*)d_in[1];
    const float* enc_w  = (const float*)d_in[2];
    const float* enc_b  = (const float*)d_in[3];
    const float* enc_g  = (const float*)d_in[4];
    const float* enc_be = (const float*)d_in[5];
    const float* erw1   = (const float*)d_in[6];
    const float* erb1   = (const float*)d_in[7];
    const float* erg1   = (const float*)d_in[8];
    const float* erbe1  = (const float*)d_in[9];
    const float* erw2   = (const float*)d_in[10];
    const float* erb2   = (const float*)d_in[11];
    const float* erg2   = (const float*)d_in[12];
    const float* erbe2  = (const float*)d_in[13];
    const float* pw1    = (const float*)d_in[14];
    const float* pb1    = (const float*)d_in[15];
    const float* pw2    = (const float*)d_in[16];
    const float* pb2    = (const float*)d_in[17];
    const float* cb     = (const float*)d_in[18];
    const float* drw1   = (const float*)d_in[19];
    const float* drb1   = (const float*)d_in[20];
    const float* drg1   = (const float*)d_in[21];
    const float* drbe1  = (const float*)d_in[22];
    const float* drw2   = (const float*)d_in[23];
    const float* drb2   = (const float*)d_in[24];
    const float* drg2   = (const float*)d_in[25];
    const float* drbe2  = (const float*)d_in[26];
    const float* dec_w  = (const float*)d_in[27];
    const float* dec_b  = (const float*)d_in[28];

    float* xrecon   = (float*)d_out;                              // [1024,768]
    float* loss_out = (float*)d_out + (size_t)1024 * 768;         // [1]
    int*   codes    = ((int*)d_out) + (size_t)1024 * 768 + 1;     // [1024,4] int32

    char* wsp = (char*)d_ws;
    size_t off = 0;
    auto alloc = [&](size_t bytes) -> char* {
        char* p = wsp + off;
        off = (off + bytes + 255) & ~(size_t)255;
        return p;
    };
    _Float16*  x16    = (_Float16*)alloc((size_t)131072 * 768 * 2);
    _Float16*  zA16   = (_Float16*)alloc((size_t)131072 * 256 * 2);
    _Float16*  zB16   = (_Float16*)alloc((size_t)131072 * 256 * 2);
    _Float16*  t16    = (_Float16*)alloc((size_t)131072 * 256 * 2);
    float*     scoresH= (float*)alloc((size_t)131072 * 128 * 4);
    _Float16*  wenc   = (_Float16*)alloc((size_t)768 * 256 * 2);
    _Float16*  wrbe   = (_Float16*)alloc((size_t)4 * 256 * 256 * 2);
    _Float16*  wpool  = (_Float16*)alloc((size_t)256 * 128 * 2);
    _Float16*  wrbd   = (_Float16*)alloc((size_t)4 * 256 * 256 * 2);
    _Float16*  wdec   = (_Float16*)alloc((size_t)256 * 768 * 2);
    _Float16*  cb16   = (_Float16*)alloc((size_t)4 * 2048 * 256 * 2);
    float*     norms  = (float*)alloc(8192 * 4);
    float*     wsm    = (float*)alloc((size_t)131072 * 4);
    float*     rbuf   = (float*)alloc((size_t)1024 * 256 * 4);
    _Float16*  r16    = (_Float16*)alloc((size_t)1024 * 256 * 2);
    float*     qout   = (float*)alloc((size_t)1024 * 256 * 4);
    _Float16*  qout16 = (_Float16*)alloc((size_t)1024 * 256 * 2);
    _Float16*  tdec16 = (_Float16*)alloc((size_t)1024 * 256 * 2);
    _Float16*  hA16   = (_Float16*)alloc((size_t)1024 * 256 * 2);
    _Float16*  hB16   = (_Float16*)alloc((size_t)1024 * 256 * 2);
    float*     Dmat   = (float*)alloc((size_t)1024 * 2048 * 4);
    int*       idxb   = (int*)alloc(1024 * 4);
    float*     vqpt   = (float*)alloc(4096 * 4);
    float*     xt     = (float*)alloc((size_t)1024 * 768 * 4);
    float*     parts  = (float*)alloc(1024 * 4);

    // ---- prep: x -> f16, weights -> f16 [N][K], codebooks f16 + norms ----
    {
        size_t n4 = (size_t)131072 * 768 / 4;
        cvt_f16_vec<<<(unsigned)((n4 + 255) / 256), 256, 0, stream>>>(x, x16, n4);
    }
    auto tc = [&](const float* s, _Float16* d, int K, int N) {
        int total = K * N;
        transpose_cvt<<<(total + 255) / 256, 256, 0, stream>>>(s, d, K, N);
    };
    tc(enc_w, wenc, 768, 256);
    tc(erw1,         wrbe,             256, 256);
    tc(erw1 + 65536, wrbe + 65536,     256, 256);
    tc(erw2,         wrbe + 2 * 65536, 256, 256);
    tc(erw2 + 65536, wrbe + 3 * 65536, 256, 256);
    tc(pw1, wpool, 256, 128);
    tc(drw1,         wrbd,             256, 256);
    tc(drw1 + 65536, wrbd + 65536,     256, 256);
    tc(drw2,         wrbd + 2 * 65536, 256, 256);
    tc(drw2 + 65536, wrbd + 3 * 65536, 256, 256);
    tc(dec_w, wdec, 256, 768);
    cvt_f16_vec<<<(4 * 2048 * 256 / 4 + 255) / 256, 256, 0, stream>>>(cb, cb16, (size_t)4 * 2048 * 256 / 4);
    cb_norms<<<32, 256, 0, stream>>>(cb, norms);

    dim3 blk(256);
    // ---- encoder (z kept in f16; LN epilogues fused) ----
    gemm_f16<32,256,2,4,1><<<dim3(1, 4096), blk, 0, stream>>>(
        x16, wenc, enc_b, enc_g, enc_be, nullptr, nullptr, zA16, 131072, 256, 768);
    gemm_f16<32,256,2,4,1><<<dim3(1, 4096), blk, 0, stream>>>(
        zA16, wrbe, erb1, erg1, erbe1, nullptr, nullptr, t16, 131072, 256, 256);
    gemm_f16<32,256,2,4,2><<<dim3(1, 4096), blk, 0, stream>>>(
        t16, wrbe + 2 * 65536, erb2, erg2, erbe2, zA16, nullptr, zB16, 131072, 256, 256);
    gemm_f16<32,256,2,4,1><<<dim3(1, 4096), blk, 0, stream>>>(
        zB16, wrbe + 65536, erb1 + 256, erg1 + 256, erbe1 + 256, nullptr, nullptr, t16, 131072, 256, 256);
    gemm_f16<32,256,2,4,2><<<dim3(1, 4096), blk, 0, stream>>>(
        t16, wrbe + 3 * 65536, erb2 + 256, erg2 + 256, erbe2 + 256, zB16, nullptr, zA16, 131072, 256, 256);

    // ---- attentive statistics pooling ----
    gemm_f16<64,128,4,2,3><<<dim3(1, 2048), blk, 0, stream>>>(
        zA16, wpool, pb1, nullptr, nullptr, nullptr, scoresH, nullptr, 131072, 128, 256);
    score_softmax<<<1024, 128, 0, stream>>>(scoresH, pw2, pb2, mask, wsm);
    pool_sum<<<1024, 256, 0, stream>>>(zA16, wsm, rbuf, r16, qout);

    // ---- residual VQ (4 codebooks) ----
    for (int i = 0; i < 4; i++) {
        gemm_f16<64,256,2,4,4><<<dim3(8, 16), blk, 0, stream>>>(
            r16, cb16 + (size_t)i * 2048 * 256, norms + i * 2048,
            nullptr, nullptr, nullptr, Dmat, nullptr, 1024, 2048, 256);
        argmin_k<<<1024, 256, 0, stream>>>(Dmat, idxb, codes, i);
        vq_update<<<1024, 256, 0, stream>>>(cb + (size_t)i * 2048 * 256, idxb,
                                            rbuf, r16, qout, qout16, vqpt + i * 1024);
    }

    // ---- decoder ----
    gemm_f16<32,256,2,4,1><<<dim3(1, 32), blk, 0, stream>>>(
        qout16, wrbd, drb1, drg1, drbe1, nullptr, nullptr, tdec16, 1024, 256, 256);
    gemm_f16<32,256,2,4,2><<<dim3(1, 32), blk, 0, stream>>>(
        tdec16, wrbd + 2 * 65536, drb2, drg2, drbe2, qout16, nullptr, hA16, 1024, 256, 256);
    gemm_f16<32,256,2,4,1><<<dim3(1, 32), blk, 0, stream>>>(
        hA16, wrbd + 65536, drb1 + 256, drg1 + 256, drbe1 + 256, nullptr, nullptr, tdec16, 1024, 256, 256);
    gemm_f16<32,256,2,4,2><<<dim3(1, 32), blk, 0, stream>>>(
        tdec16, wrbd + 3 * 65536, drb2 + 256, drg2 + 256, drbe2 + 256, hA16, nullptr, hB16, 1024, 256, 256);
    gemm_f16<64,256,2,4,0><<<dim3(3, 16), blk, 0, stream>>>(
        hB16, wdec, dec_b, nullptr, nullptr, nullptr, xrecon, nullptr, 1024, 768, 256);

    // ---- reconstruction target + total loss ----
    x_target_k<<<1024, 256, 0, stream>>>(x, mask, xt);
    recon_part_k<<<1024, 256, 0, stream>>>(xrecon, xt, parts);
    final_loss_k<<<1, 256, 0, stream>>>(parts, vqpt, loss_out);
}
